// FP8DynamicLinear_18004502905012
// MI455X (gfx1250) — 8-bit matrix operand order repaired and checked by reading, compiled, not run on hardware
//
#include <hip/hip_runtime.h>
#include <hip/hip_bf16.h>

typedef __attribute__((ext_vector_type(16))) int   v16i;
typedef __attribute__((ext_vector_type(8)))  float v8f;
typedef int v4i_vs __attribute__((vector_size(16)));

#define FP8_MAX 448.0f
#define BLK 128
#define LDS_STRIDE 144

#define AS_GLOBAL __attribute__((address_space(1)))
#define AS_LDS    __attribute__((address_space(3)))

#if __has_builtin(__builtin_amdgcn_global_load_async_to_lds_b128)
#define USE_ASYNC_LDS 1
#else
#define USE_ASYNC_LDS 0
#endif

#if __has_builtin(__builtin_amdgcn_s_wait_asynccnt)
#define WAIT_ASYNC0() __builtin_amdgcn_s_wait_asynccnt(0)
#else
#define WAIT_ASYNC0() asm volatile("s_wait_asynccnt 0x0" ::: "memory")
#endif

__device__ __forceinline__ unsigned int f32_to_e4m3(float f) {
    unsigned int sign = (__float_as_uint(f) >> 24) & 0x80u;
    float a = fabsf(f);
    a = fminf(a, FP8_MAX);
    unsigned int code;
    if (a < 0.015625f) {
        int q = __float2int_rn(a * 512.0f);
        code = (unsigned int)q;
    } else {
        unsigned int au = __float_as_uint(a);
        au += 0x7FFFFu + ((au >> 20) & 1u);
        int e = (int)(au >> 23) - 127;
        code = ((unsigned int)(e + 7) << 3) | ((au >> 20) & 7u);
        if (code > 0x7Eu) code = 0x7Eu;
    }
    return sign | code;
}

__device__ __forceinline__ float bf16_round(float x) {
    return __bfloat162float(__float2bfloat16(x));
}

__global__ void init_amax_kernel(unsigned int* amaxBits) {
    if (threadIdx.x == 0 && blockIdx.x == 0) amaxBits[0] = 0u;
}

__global__ __launch_bounds__(256)
void amax_kernel(const float* __restrict__ x, unsigned int* __restrict__ amaxBits, long n4) {
    float m = 0.0f;
    const float4* x4 = (const float4*)x;
    long stride = (long)gridDim.x * blockDim.x;
    for (long i = (long)blockIdx.x * blockDim.x + threadIdx.x; i < n4; i += stride) {
        float4 v = x4[i];
        m = fmaxf(m, fabsf(bf16_round(v.x)));
        m = fmaxf(m, fabsf(bf16_round(v.y)));
        m = fmaxf(m, fabsf(bf16_round(v.z)));
        m = fmaxf(m, fabsf(bf16_round(v.w)));
    }
    #pragma unroll
    for (int off = 16; off > 0; off >>= 1)
        m = fmaxf(m, __shfl_xor(m, off, 32));
    __shared__ float sm[8];
    if ((threadIdx.x & 31) == 0) sm[threadIdx.x >> 5] = m;
    __syncthreads();
    if (threadIdx.x < 8) {
        m = sm[threadIdx.x];
        #pragma unroll
        for (int off = 4; off > 0; off >>= 1)
            m = fmaxf(m, __shfl_xor(m, off, 32));
        if (threadIdx.x == 0) atomicMax(amaxBits, __float_as_uint(m));
    }
}

__global__ __launch_bounds__(256)
void quant_x_kernel(const float* __restrict__ x, const unsigned int* __restrict__ amaxBits,
                    unsigned int* __restrict__ qx4, long n4) {
    const float scale = fmaxf(__uint_as_float(amaxBits[0]), 1e-12f) / FP8_MAX;
    const float4* x4 = (const float4*)x;
    long stride = (long)gridDim.x * blockDim.x;
    for (long i = (long)blockIdx.x * blockDim.x + threadIdx.x; i < n4; i += stride) {
        float4 v = x4[i];
        unsigned int b0 = f32_to_e4m3(bf16_round(v.x) / scale);
        unsigned int b1 = f32_to_e4m3(bf16_round(v.y) / scale);
        unsigned int b2 = f32_to_e4m3(bf16_round(v.z) / scale);
        unsigned int b3 = f32_to_e4m3(bf16_round(v.w) / scale);
        qx4[i] = b0 | (b1 << 8) | (b2 << 16) | (b3 << 24);
    }
}

__global__ __launch_bounds__(256)
void quant_w_kernel(const float* __restrict__ w, unsigned int* __restrict__ qw4, long n4) {
    const float4* w4 = (const float4*)w;
    long stride = (long)gridDim.x * blockDim.x;
    for (long i = (long)blockIdx.x * blockDim.x + threadIdx.x; i < n4; i += stride) {
        float4 v = w4[i];
        unsigned int b0 = f32_to_e4m3(v.x);
        unsigned int b1 = f32_to_e4m3(v.y);
        unsigned int b2 = f32_to_e4m3(v.z);
        unsigned int b3 = f32_to_e4m3(v.w);
        qw4[i] = b0 | (b1 << 8) | (b2 << 16) | (b3 << 24);
    }
}

__device__ __forceinline__ v16i load_a_frag(const unsigned char* sA, int row, int half) {
    v16i r;
    const unsigned char* p = sA + row * LDS_STRIDE + half * 8;
    #pragma unroll
    for (int i = 0; i < 8; ++i) {
        uint2 d = *(const uint2*)(p + 16 * i);
        r[2 * i]     = (int)d.x;
        r[2 * i + 1] = (int)d.y;
    }
    return r;
}

__device__ __forceinline__ v16i load_b_frag(const unsigned char* sB, int row, int half) {
    v16i r;
    const unsigned char* p = sB + row * LDS_STRIDE + half * 8;
    #pragma unroll
    for (int j = 0; j < 4; ++j) {
        uint2 d = *(const uint2*)(p + 32 * j);
        uint2 e = *(const uint2*)(p + 32 * j + 16);
        r[4 * j]     = (int)d.x;
        r[4 * j + 1] = (int)d.y;
        r[4 * j + 2] = (int)e.x;
        r[4 * j + 3] = (int)e.y;
    }
    return r;
}

__global__ __launch_bounds__(256)
void fp8_gemm_kernel(const unsigned char* __restrict__ qx,
                     const unsigned char* __restrict__ qw,
                     const unsigned int* __restrict__ amaxBits,
                     const float* __restrict__ wScale,
                     const float* __restrict__ bias,
                     __hip_bfloat16* __restrict__ out,
                     int M, int N, int K) {
    __shared__ unsigned char sA[2][BLK * LDS_STRIDE];
    __shared__ unsigned char sB[2][BLK * LDS_STRIDE];

    const int tid   = threadIdx.x;
    const int lane  = tid & 31;
    const int wave  = tid >> 5;
    const int waveM = wave >> 1;
    const int waveN = wave & 1;
    const int half  = lane >> 4;
    const int rl    = lane & 15;

    const int mBase = blockIdx.y * BLK;
    const int nBase = blockIdx.x * BLK;

    v8f acc[2][4];
    #pragma unroll
    for (int mi = 0; mi < 2; ++mi)
        #pragma unroll
        for (int ni = 0; ni < 4; ++ni)
            acc[mi][ni] = (v8f){0.f, 0.f, 0.f, 0.f, 0.f, 0.f, 0.f, 0.f};

    const int ldRow = tid >> 3;
    const int ldCol = (tid & 7) * 16;

    const int nk = K / BLK;

#if USE_ASYNC_LDS
    #define ISSUE_TILE(kt, buf)                                                          \
        do {                                                                             \
            _Pragma("unroll")                                                            \
            for (int i = 0; i < 4; ++i) {                                                \
                const int r = ldRow + 32 * i;                                            \
                unsigned char* ga = const_cast<unsigned char*>(                          \
                    qx + (size_t)(mBase + r) * K + (kt) + ldCol);                        \
                unsigned char* gb = const_cast<unsigned char*>(                          \
                    qw + (size_t)(nBase + r) * K + (kt) + ldCol);                        \
                __builtin_amdgcn_global_load_async_to_lds_b128(                          \
                    (AS_GLOBAL v4i_vs*)ga,                                               \
                    (AS_LDS v4i_vs*)&sA[buf][r * LDS_STRIDE + ldCol], 0, 0);             \
                __builtin_amdgcn_global_load_async_to_lds_b128(                          \
                    (AS_GLOBAL v4i_vs*)gb,                                               \
                    (AS_LDS v4i_vs*)&sB[buf][r * LDS_STRIDE + ldCol], 0, 0);             \
            }                                                                            \
        } while (0)

    ISSUE_TILE(0, 0);
    for (int k = 0; k < nk; ++k) {
        const int cur = k & 1;
        WAIT_ASYNC0();
        __syncthreads();
        if (k + 1 < nk)
            ISSUE_TILE((k + 1) * BLK, cur ^ 1);
#else
    uint4 ra[4], rb[4];
    #pragma unroll
    for (int i = 0; i < 4; ++i) {
        const int r = ldRow + 32 * i;
        ra[i] = *(const uint4*)(qx + (size_t)(mBase + r) * K + ldCol);
        rb[i] = *(const uint4*)(qw + (size_t)(nBase + r) * K + ldCol);
    }
    for (int k = 0; k < nk; ++k) {
        const int cur = k & 1;
        #pragma unroll
        for (int i = 0; i < 4; ++i) {
            const int r = ldRow + 32 * i;
            *(uint4*)(sA[cur] + r * LDS_STRIDE + ldCol) = ra[i];
            *(uint4*)(sB[cur] + r * LDS_STRIDE + ldCol) = rb[i];
        }
        if (k + 1 < nk) {
            const int kt = (k + 1) * BLK;
            #pragma unroll
            for (int i = 0; i < 4; ++i) {
                const int r = ldRow + 32 * i;
                ra[i] = *(const uint4*)(qx + (size_t)(mBase + r) * K + kt + ldCol);
                rb[i] = *(const uint4*)(qw + (size_t)(nBase + r) * K + kt + ldCol);
            }
        }
        __syncthreads();
#endif

        v16i af[2], bfrag[4];
        #pragma unroll
        for (int mi = 0; mi < 2; ++mi)
            af[mi] = load_a_frag(sA[cur], waveM * 32 + mi * 16 + rl, half);
        #pragma unroll
        for (int ni = 0; ni < 4; ++ni)
            bfrag[ni] = load_b_frag(sB[cur], waveN * 64 + ni * 16 + rl, half);

        #pragma unroll
        for (int mi = 0; mi < 2; ++mi)
            #pragma unroll
            for (int ni = 0; ni < 4; ++ni)
                acc[mi][ni] = __builtin_amdgcn_wmma_f32_16x16x128_fp8_fp8(
                    af[mi], bfrag[ni], (short)0, acc[mi][ni], false, false);
    }

    const float xScale = fmaxf(__uint_as_float(amaxBits[0]), 1e-12f) / FP8_MAX;
    const float cs = xScale * wScale[0];
    #pragma unroll
    for (int ni = 0; ni < 4; ++ni) {
        const int n = nBase + waveN * 64 + ni * 16 + rl;
        const float bv = bias[n];
        #pragma unroll
        for (int mi = 0; mi < 2; ++mi) {
            #pragma unroll
            for (int v = 0; v < 8; ++v) {
                const int m = mBase + waveM * 32 + mi * 16 + half * 8 + v;
                out[(size_t)m * N + n] = __float2bfloat16(acc[mi][ni][v] * cs + bv);
            }
        }
    }
}

extern "C" void kernel_launch(void* const* d_in, const int* in_sizes, int n_in,
                              void* d_out, int out_size, void* d_ws, size_t ws_size,
                              hipStream_t stream) {
    const float* x      = (const float*)d_in[0];
    const float* w      = (const float*)d_in[1];
    const float* wScale = (const float*)d_in[2];
    const float* bias   = (const float*)d_in[3];
    __hip_bfloat16* out = (__hip_bfloat16*)d_out;

    const long N = in_sizes[3];
    const long K = (long)in_sizes[1] / N;
    const long M = (long)in_sizes[0] / K;

    unsigned int*  amaxBits = (unsigned int*)d_ws;
    unsigned char* qx = (unsigned char*)d_ws + 256;
    unsigned char* qw = qx + (size_t)M * K;

    init_amax_kernel<<<1, 32, 0, stream>>>(amaxBits);
    amax_kernel<<<1024, 256, 0, stream>>>(x, amaxBits, M * K / 4);
    quant_x_kernel<<<2048, 256, 0, stream>>>(x, amaxBits, (unsigned int*)qx, M * K / 4);
    quant_w_kernel<<<1024, 256, 0, stream>>>(w, (unsigned int*)qw, N * K / 4);

    dim3 grid((unsigned)(N / BLK), (unsigned)(M / BLK));
    fp8_gemm_kernel<<<grid, 256, 0, stream>>>(qx, qw, amaxBits, wScale, bias, out,
                                              (int)M, (int)N, (int)K);
}
